// GIN_57260503991115
// MI455X (gfx1250) — compile-verified, matrix operand order checked by reading, not run on hardware
//
#include <hip/hip_runtime.h>
#include <hip/hip_bf16.h>

typedef __attribute__((ext_vector_type(16))) _Float16 v16h;
typedef __attribute__((ext_vector_type(8)))  float    v8f;

#define GIN_N 100000
#define GIN_E 1200000
#define GIN_G 1000

// ---------------------------------------------------------------------------
// Fragment layouts (wave32 WMMA 16x16x32 f16, CDNA5 ISA 7.12.2)
// A (16x32 f16): lane L holds row M=L&15; khb = (L<16 ? 0 : 8);
//   halfs j0..7  = K kstart+khb .. +7,  j8..15 = K kstart+16+khb .. +7.
// B (32x16 f16): symmetric; col = L&15, same K striping.
// C/D (16x16 f32): col = L&15, VGPR r -> row = r + 8*(L>=16).
// ---------------------------------------------------------------------------

// A-fragment from a row-major f32 row. Group validity is uniform per 8-half
// group for our shapes (KVALID in {16,64}, kstart in {0,32}), so the guards
// constant-fold after unrolling and each group is two float4 loads.
template <int KVALID>
__device__ inline v16h frag_a_global(const float* __restrict__ rowptr, bool valid,
                                     int kstart, int lane) {
  const int khb = (lane & 16) ? 8 : 0;
  v16h f = {};
  if (valid) {
    if (kstart + 15 < KVALID) {           // low 8 halfs: K = kstart+khb .. +7
      const float4 p0 = *(const float4*)(rowptr + kstart + khb);
      const float4 p1 = *(const float4*)(rowptr + kstart + khb + 4);
      f[0] = (_Float16)p0.x; f[1] = (_Float16)p0.y;
      f[2] = (_Float16)p0.z; f[3] = (_Float16)p0.w;
      f[4] = (_Float16)p1.x; f[5] = (_Float16)p1.y;
      f[6] = (_Float16)p1.z; f[7] = (_Float16)p1.w;
    }
    if (kstart + 31 < KVALID) {           // high 8 halfs: K = kstart+16+khb .. +7
      const float4 q0 = *(const float4*)(rowptr + kstart + 16 + khb);
      const float4 q1 = *(const float4*)(rowptr + kstart + 16 + khb + 4);
      f[8]  = (_Float16)q0.x; f[9]  = (_Float16)q0.y;
      f[10] = (_Float16)q0.z; f[11] = (_Float16)q0.w;
      f[12] = (_Float16)q1.x; f[13] = (_Float16)q1.y;
      f[14] = (_Float16)q1.z; f[15] = (_Float16)q1.w;
    }
  }
  return f;
}

// Fragment from LDS: colbase points at a contiguous run of K halfs (one column
// of W^T, or one row of the h1 tile) -> two contiguous 16B ds_load_b128 groups.
__device__ inline v16h frag_from_lds(const _Float16* colbase, int kstart, int lane) {
  const int khb = (lane & 16) ? 8 : 0;
  v16h f;
#pragma unroll
  for (int j = 0; j < 8; ++j) {
    f[j]     = colbase[kstart + khb + j];
    f[j + 8] = colbase[kstart + 16 + khb + j];
  }
  return f;
}

// ---------------------------------------------------------------------------
// Fused GIN MLP: h_out = relu( relu(BN(agg @ wa + ba)) @ wb + bb )
// KT = K=32 WMMA steps of GEMM1 (1 for layer 1, 2 else); KVALID = agg row
// width (16 for layer 1 padded, 64 else). Block = 256 threads = 8 wave32s;
// each wave owns a 16-row tile (128 rows per block).
// ---------------------------------------------------------------------------
template <int KT, int KVALID>
__global__ __launch_bounds__(256) void gin_mlp_kernel(
    const float* __restrict__ agg,
    const float* __restrict__ wa, int kw,
    const float* __restrict__ ba,
    const float* __restrict__ gma, const float* __restrict__ bta,
    const float* __restrict__ mu,  const float* __restrict__ var,
    const float* __restrict__ wb,  const float* __restrict__ bb,
    float* __restrict__ hout, int nrows) {
  constexpr int K = KT * 32;
  __shared__ _Float16 sWA[64][K];        // wa transposed: [col][k]
  __shared__ _Float16 sWB[64][64];       // wb transposed: [col][k]
  __shared__ _Float16 sH1[8][16][64];    // per-wave intermediate, row-major
  __shared__ float sScale[64], sShift[64], sBias1[64], sBias2[64];

  const int tid  = threadIdx.x;
  const int lane = tid & 31;
  const int wv   = tid >> 5;

  // Folded BN affine + biases, once per block.
  if (tid < 64) {
    const float s = gma[tid] * rsqrtf(var[tid] + 1e-5f);
    sScale[tid] = s;
    sShift[tid] = bta[tid] - mu[tid] * s;
    sBias1[tid] = ba[tid];
    sBias2[tid] = bb[tid];
  }
  // Cooperative weight staging (f32 -> f16, transposed), zero-pad wa rows >= kw.
  for (int idx = tid; idx < 64 * K; idx += 256) {
    const int col = idx / K;
    const int k   = idx - col * K;
    sWA[col][k] = (k < kw) ? (_Float16)wa[k * 64 + col] : (_Float16)0.0f;
  }
  for (int idx = tid; idx < 64 * 64; idx += 256) {
    const int col = idx >> 6;
    const int k   = idx & 63;
    sWB[col][k] = (_Float16)wb[k * 64 + col];
  }
  __syncthreads();

  const int rowbase = (blockIdx.x * 8 + wv) * 16;
  const int mrow    = rowbase + (lane & 15);
  const bool valid  = (mrow < nrows);
  const float* rowptr = agg + (size_t)mrow * KVALID;

  // ---- GEMM1: agg[16xK] @ wa[Kx64] ----
  v8f acc[4] = {};
#pragma unroll
  for (int kt = 0; kt < KT; ++kt) {
    const v16h a = frag_a_global<KVALID>(rowptr, valid, kt * 32, lane);
#pragma unroll
    for (int nt = 0; nt < 4; ++nt) {
      const v16h b = frag_from_lds(&sWA[nt * 16 + (lane & 15)][0], kt * 32, lane);
      acc[nt] = __builtin_amdgcn_wmma_f32_16x16x32_f16(
          false, a, false, b, (short)0, acc[nt], false, false);
    }
  }

  // ---- bias + BN(eval) + ReLU, stage h1 tile to LDS as f16 ----
  const int rbase = (lane & 16) ? 8 : 0;
#pragma unroll
  for (int nt = 0; nt < 4; ++nt) {
    const int col = nt * 16 + (lane & 15);
    const float s    = sScale[col];
    const float bsh  = sShift[col];
    const float bias = sBias1[col];
#pragma unroll
    for (int r = 0; r < 8; ++r) {
      const float y = fmaxf((acc[nt][r] + bias) * s + bsh, 0.0f);
      sH1[wv][rbase + r][col] = (_Float16)y;
    }
  }
  __syncthreads();

  // ---- GEMM2: h1[16x64] @ wb[64x64] ----
  v8f acc2[4] = {};
#pragma unroll
  for (int kt = 0; kt < 2; ++kt) {
    const v16h a = frag_from_lds(&sH1[wv][lane & 15][0], kt * 32, lane);
#pragma unroll
    for (int nt = 0; nt < 4; ++nt) {
      const v16h b = frag_from_lds(&sWB[nt * 16 + (lane & 15)][0], kt * 32, lane);
      acc2[nt] = __builtin_amdgcn_wmma_f32_16x16x32_f16(
          false, a, false, b, (short)0, acc2[nt], false, false);
    }
  }

  // ---- bias + ReLU, store f32 ----
#pragma unroll
  for (int nt = 0; nt < 4; ++nt) {
    const int col = nt * 16 + (lane & 15);
    const float bias = sBias2[col];
#pragma unroll
    for (int r = 0; r < 8; ++r) {
      const int grow = rowbase + rbase + r;
      if (grow < nrows)
        hout[(size_t)grow * 64 + col] = fmaxf(acc2[nt][r] + bias, 0.0f);
    }
  }
}

// ---------------------------------------------------------------------------
// Aggregation / pooling / head kernels (L2-resident atomic traffic)
// ---------------------------------------------------------------------------

// agg[N x 16] = x[N x 11] zero-padded (layer-1 self term + K padding)
__global__ void pad_x_kernel(const float* __restrict__ x, float* __restrict__ agg, int n) {
  const int tid = blockIdx.x * blockDim.x + threadIdx.x;
  if (tid >= n * 16) return;
  const int i = tid >> 4, c = tid & 15;
  agg[tid] = (c < 11) ? x[i * 11 + c] : 0.0f;
}

// agg[dst] += x[src] for 11-dim layer-1 features
__global__ void scatter11_kernel(const int* __restrict__ ei, const float* __restrict__ x,
                                 float* __restrict__ agg, int e) {
  const int t = blockIdx.x * blockDim.x + threadIdx.x;
  if (t >= e) return;
  __builtin_prefetch(ei + t + 2048, 0, 1);          // global_prefetch_b8
  const int s = ei[t], d = ei[e + t];
  const float* sp = x + (size_t)s * 11;
  float* dp = agg + (size_t)d * 16;
#pragma unroll
  for (int c = 0; c < 11; ++c) atomicAdd(dp + c, sp[c]);
}

// agg[dst] += h[src], 64-dim, float4 per thread
__global__ void scatter64_kernel(const int* __restrict__ ei, const float* __restrict__ h,
                                 float* __restrict__ agg, int e) {
  const long t = (long)blockIdx.x * blockDim.x + threadIdx.x;
  if (t >= (long)e * 16) return;
  const int eidx = (int)(t >> 4);
  const int q = ((int)t & 15) * 4;
  __builtin_prefetch(ei + eidx + 512, 0, 1);        // global_prefetch_b8
  const int s = ei[eidx], d = ei[e + eidx];
  const float4 vsrc = *(const float4*)(h + (size_t)s * 64 + q);
  float* dp = agg + (size_t)d * 64 + q;
  atomicAdd(dp + 0, vsrc.x);
  atomicAdd(dp + 1, vsrc.y);
  atomicAdd(dp + 2, vsrc.z);
  atomicAdd(dp + 3, vsrc.w);
}

__global__ void copy_kernel(const float* __restrict__ a, float* __restrict__ b, long n4) {
  const long t = (long)blockIdx.x * blockDim.x + threadIdx.x;
  if (t < n4) ((float4*)b)[t] = ((const float4*)a)[t];
}

__global__ void zero_kernel(float* __restrict__ b, long n) {
  const long t = (long)blockIdx.x * blockDim.x + threadIdx.x;
  if (t < n) b[t] = 0.0f;
}

// hg[batch[i]] += h[i]  (global_add_pool)
__global__ void pool_kernel(const float* __restrict__ h, const int* __restrict__ batch,
                            float* __restrict__ hg, int n) {
  const int t = blockIdx.x * blockDim.x + threadIdx.x;
  if (t >= n * 16) return;
  const int i = t >> 4, q = (t & 15) * 4;
  const int b = batch[i];
  const float4 vsrc = *(const float4*)(h + (size_t)i * 64 + q);
  float* dp = hg + (size_t)b * 64 + q;
  atomicAdd(dp + 0, vsrc.x);
  atomicAdd(dp + 1, vsrc.y);
  atomicAdd(dp + 2, vsrc.z);
  atomicAdd(dp + 3, vsrc.w);
}

// out[g] = relu(hg[g] @ lw1 + lb1) @ lw2 + lb2
__global__ void head_kernel(const float* __restrict__ hg, const float* __restrict__ lw1,
                            const float* __restrict__ lb1, const float* __restrict__ lw2,
                            const float* __restrict__ lb2, float* __restrict__ out, int g) {
  __shared__ float s[64];
  const int gi = blockIdx.x, t = threadIdx.x;
  const float* row = hg + (size_t)gi * 64;
  float acc = lb1[t];
#pragma unroll 8
  for (int k = 0; k < 64; ++k) acc += row[k] * lw1[k * 64 + t];
  s[t] = fmaxf(acc, 0.0f);
  __syncthreads();
  if (t == 0) {
    float o = lb2[0];
#pragma unroll 8
    for (int k = 0; k < 64; ++k) o += s[k] * lw2[k];
    out[gi] = o;
  }
}

// ---------------------------------------------------------------------------

extern "C" void kernel_launch(void* const* d_in, const int* in_sizes, int n_in,
                              void* d_out, int out_size, void* d_ws, size_t ws_size,
                              hipStream_t stream) {
  (void)in_sizes; (void)n_in; (void)out_size; (void)ws_size;
  const int N = GIN_N, E = GIN_E, G = GIN_G;

  const float* x     = (const float*)d_in[0];
  const int*   ei    = (const int*)d_in[1];
  const int*   batch = (const int*)d_in[2];
  auto P = [&](int i) { return (const float*)d_in[i]; };

  float* agg = (float*)d_ws;                 // N*64 f32
  float* h   = agg + (size_t)N * 64;         // N*64 f32
  float* hg  = h + (size_t)N * 64;           // G*64 f32

  const int thr = 256;

  // ---- layer 1 (K=11 padded to 16 cols, one K=32 WMMA step) ----
  pad_x_kernel<<<(N * 16 + thr - 1) / thr, thr, 0, stream>>>(x, agg, N);
  scatter11_kernel<<<(E + thr - 1) / thr, thr, 0, stream>>>(ei, x, agg, E);
  gin_mlp_kernel<1, 16><<<(N + 127) / 128, 256, 0, stream>>>(
      agg, P(3), 11, P(4), P(5), P(6), P(7), P(8), P(9), P(10), h, N);

  // ---- layers 2, 3 (K=64) ----
  for (int l = 0; l < 2; ++l) {
    const int o = 11 + l * 8;
    copy_kernel<<<(int)(((long)N * 16 + thr - 1) / thr), thr, 0, stream>>>(h, agg, (long)N * 16);
    scatter64_kernel<<<(int)(((long)E * 16 + thr - 1) / thr), thr, 0, stream>>>(ei, h, agg, E);
    gin_mlp_kernel<2, 64><<<(N + 127) / 128, 256, 0, stream>>>(
        agg, P(o), 64, P(o + 1), P(o + 2), P(o + 3), P(o + 4), P(o + 5),
        P(o + 6), P(o + 7), h, N);
  }

  // ---- global_add_pool + head ----
  zero_kernel<<<(G * 64 + thr - 1) / thr, thr, 0, stream>>>(hg, (long)G * 64);
  pool_kernel<<<(N * 16 + thr - 1) / thr, thr, 0, stream>>>(h, batch, hg, N);
  head_kernel<<<G, 64, 0, stream>>>(hg, P(27), P(28), P(29), P(30), (float*)d_out, G);
}
